// MambaExpert_13331578487396
// MI455X (gfx1250) — compile-verified, matrix operand order checked by reading, not run on hardware
//
#include <hip/hip_runtime.h>
#include <math.h>

#define D_MODEL 1024
#define D_STATE 16
#define D_INNER 2048
#define DT_RANK 64
#define BATCH   4
#define SEQ     2048
#define MROWS   (BATCH * SEQ)            // 8192 rows for every GEMM
#define DBL_N   (DT_RANK + 2 * D_STATE)  // 96

// ---------- types ----------
typedef __attribute__((ext_vector_type(16))) __bf16  v16bf;
typedef __attribute__((ext_vector_type(8)))  float   v8f;
typedef __attribute__((ext_vector_type(4)))  unsigned int u32x4;

union FragU { u32x4 q[2]; v16bf v; };

__device__ __forceinline__ unsigned short f32_to_bf16(float f) {
  unsigned int x = __float_as_uint(f);
  x += 0x7FFFu + ((x >> 16) & 1u);          // round-to-nearest-even
  return (unsigned short)(x >> 16);
}
__device__ __forceinline__ float bf16_to_f32(unsigned short h) {
  return __uint_as_float(((unsigned int)h) << 16);
}

// ---------- CDNA5 async global->LDS (ASYNCcnt path, no VGPR staging) ----------
// LDS byte address = low 32 bits of the generic shared pointer (ISA 10.2:
// applies to addresses in the shared aperture).
__device__ __forceinline__ void async_load_b128(unsigned lds_addr, const void* gptr) {
  asm volatile("global_load_async_to_lds_b128 %0, %1, off"
               :: "v"(lds_addr), "v"((unsigned long long)(size_t)gptr)
               : "memory");
}
__device__ __forceinline__ void wait_async0() {
  asm volatile("s_wait_asynccnt 0" ::: "memory");
}

// ---------- f32 -> bf16 conversion ----------
__global__ __launch_bounds__(256) void cvt_f32_bf16(const float* __restrict__ in,
                                                    unsigned short* __restrict__ out,
                                                    int n) {
  int i = blockIdx.x * 256 + threadIdx.x;
  if (i < n) out[i] = f32_to_bf16(in[i]);
}

// ---------- tiled bf16 WMMA GEMM:  C[M,N] = A[M,K] * W[N,K]^T ----------
// Block tile 128x128, K-step 32, 8 waves in 4(m) x 2(n); wave tile 32x64
// (2x4 WMMA accumulators). Global->LDS staging via async b128 loads.
#define BM  128
#define BN  128
#define KS  32
#define LDT 40   // LDS row stride (halves): 80 B rows keep 16-B alignment

__global__ __launch_bounds__(256) void gemm_bf16(
    const unsigned short* __restrict__ A, int lda,
    const unsigned short* __restrict__ W, int ldw,
    void* __restrict__ Cout, int N, int K, int ldc, int out_bf16)
{
  __shared__ __align__(16) unsigned short sA[BM * LDT];
  __shared__ __align__(16) unsigned short sB[BN * LDT];

  const int t     = threadIdx.x;
  const int lane  = t & 31;
  const int wv    = t >> 5;            // wave 0..7
  const int wm    = (wv >> 1) * 32;    // wave tile row in block tile (0..96)
  const int wn    = (wv & 1) * 64;     // wave tile col in block tile (0,64)
  const int lrow  = lane & 15;
  const int khalf = (lane >> 4) * 16;  // K half selected by lane group
  const int m0 = blockIdx.y * BM;
  const int n0 = blockIdx.x * BN;

  // cooperative staging: each thread owns 32 B of A tile and 32 B of B tile
  const int arow = t >> 1, acol = (t & 1) * 16;   // 128 rows x 32 halves
  const int brow = t >> 1, bcol = (t & 1) * 16;   // 128 rows x 32 halves
  int bgn = n0 + brow; if (bgn >= N) bgn = N - 1; // clamp (stores are masked)

  const unsigned ldsA0 = (unsigned)(unsigned long long)(const void*)(sA + arow * LDT + acol);
  const unsigned ldsB0 = (unsigned)(unsigned long long)(const void*)(sB + brow * LDT + bcol);
  const unsigned short* gA = A + (size_t)(m0 + arow) * lda + acol;
  const unsigned short* gB = W + (size_t)bgn * ldw + bcol;

  const v8f vzero = {0.f,0.f,0.f,0.f,0.f,0.f,0.f,0.f};
  v8f acc[2][4];
#pragma unroll
  for (int i = 0; i < 2; ++i)
#pragma unroll
    for (int j = 0; j < 4; ++j) acc[i][j] = vzero;

  for (int kb = 0; kb < K; kb += KS, gA += KS, gB += KS) {
    __syncthreads();                       // tiles free to overwrite
    async_load_b128(ldsA0,      gA);
    async_load_b128(ldsA0 + 16, gA + 8);
    async_load_b128(ldsB0,      gB);
    async_load_b128(ldsB0 + 16, gB + 8);
    if (kb + KS < K) {                     // hint next slab -> global_prefetch_b8
      __builtin_prefetch(gA + KS, 0, 1);
      __builtin_prefetch(gB + KS, 0, 1);
    }
    wait_async0();                         // this wave's async LDS writes done
    __syncthreads();                       // all waves' writes visible

    FragU af[2], bfv[4];
#pragma unroll
    for (int i = 0; i < 2; ++i) {
      const u32x4* pa = (const u32x4*)(sA + (wm + i*16 + lrow) * LDT + khalf);
      af[i].q[0] = pa[0]; af[i].q[1] = pa[1];
    }
#pragma unroll
    for (int j = 0; j < 4; ++j) {
      const u32x4* pb = (const u32x4*)(sB + (wn + j*16 + lrow) * LDT + khalf);
      bfv[j].q[0] = pb[0]; bfv[j].q[1] = pb[1];
    }
#pragma unroll
    for (int i = 0; i < 2; ++i)
#pragma unroll
      for (int j = 0; j < 4; ++j)
        acc[i][j] = __builtin_amdgcn_wmma_f32_16x16x32_bf16(
            false, af[i].v, false, bfv[j].v, (short)0, acc[i][j], false, false);
  }

  // C/D layout: lanes 0-15 -> M=r, lanes 16-31 -> M=r+8 ; N = lane&15
  const int mloc = (lane >> 4) * 8;
#pragma unroll
  for (int i = 0; i < 2; ++i)
#pragma unroll
    for (int j = 0; j < 4; ++j) {
      int colg = n0 + wn + j*16 + lrow;
      if (colg < N) {
#pragma unroll
        for (int r = 0; r < 8; ++r) {
          int rowg = m0 + wm + i*16 + mloc + r;
          float v = acc[i][j][r];
          if (out_bf16)
            ((unsigned short*)Cout)[(size_t)rowg * ldc + colg] = f32_to_bf16(v);
          else
            ((float*)Cout)[(size_t)rowg * ldc + colg] = v;
        }
      }
    }
}

// ---------- depthwise conv (K=4, causal) + bias + SiLU ----------
__global__ __launch_bounds__(256) void conv_silu(
    const unsigned short* __restrict__ xz,   // [B,L,2*D_INNER], xi = first half
    const float* __restrict__ cw,            // [D_INNER,4]
    const float* __restrict__ cb,            // [D_INNER]
    unsigned short* __restrict__ u)          // [B,L,D_INNER]
{
  int idx = blockIdx.x * 256 + threadIdx.x;       // over B*L*D_INNER
  int c = idx & (D_INNER - 1);
  int l = (idx >> 11) & (SEQ - 1);
  int b = idx >> 22;
  float acc = cb[c];
  float w0 = cw[c*4+0], w1 = cw[c*4+1], w2 = cw[c*4+2], w3 = cw[c*4+3];
  size_t rbase = ((size_t)b * SEQ + l) * (2 * D_INNER) + c;
  const size_t stride = 2 * D_INNER;
  acc += bf16_to_f32(xz[rbase]) * w3;
  if (l >= 1) acc += bf16_to_f32(xz[rbase - stride])     * w2;
  if (l >= 2) acc += bf16_to_f32(xz[rbase - 2 * stride]) * w1;
  if (l >= 3) acc += bf16_to_f32(xz[rbase - 3 * stride]) * w0;
  float s = acc / (1.f + __expf(-acc));
  u[idx] = f32_to_bf16(s);
}

// ---------- dt = softplus(dt_pre + b_dt) ----------
__global__ __launch_bounds__(256) void dt_softplus(
    const unsigned short* __restrict__ dtp,  // [B,L,D_INNER]
    const float* __restrict__ bdt,           // [D_INNER]
    unsigned short* __restrict__ dt)
{
  int idx = blockIdx.x * 256 + threadIdx.x;
  int c = idx & (D_INNER - 1);
  float v = bf16_to_f32(dtp[idx]) + bdt[c];
  float sp = (v > 20.f) ? v : log1pf(__expf(v));
  dt[idx] = f32_to_bf16(sp);
}

// ---------- selective scan: thread = (channel, state), shuffle-reduce y ----------
__global__ __launch_bounds__(256) void scan_kernel(
    const unsigned short* __restrict__ dt_bf,   // [B,L,D_INNER]
    const unsigned short* __restrict__ u_bf,    // [B,L,D_INNER]
    const unsigned short* __restrict__ dbl_bf,  // [B,L,96] (B at 64, C at 80)
    const unsigned short* __restrict__ xz_bf,   // [B,L,4096] (z at +2048)
    const float* __restrict__ A_log,            // [D_INNER,16]
    const float* __restrict__ D_skip,           // [D_INNER]
    unsigned short* __restrict__ y_bf)          // [B,L,D_INNER]
{
  const int t  = threadIdx.x;
  const int s  = t & 15;                       // state index
  const int ch = blockIdx.x * 16 + (t >> 4);   // channel
  const int b  = blockIdx.y;
  const float Arow = -__expf(A_log[ch * D_STATE + s]);
  const float Dsk  = D_skip[ch];
  const size_t base = (size_t)b * SEQ;
  float h = 0.f;
  for (int tt = 0; tt < SEQ; ++tt) {
    size_t r  = base + tt;
    float dt = bf16_to_f32(dt_bf[r * D_INNER + ch]);   // lane-uniform in group
    float uu = bf16_to_f32(u_bf [r * D_INNER + ch]);
    float Bv = bf16_to_f32(dbl_bf[r * DBL_N + DT_RANK + s]);
    float Cv = bf16_to_f32(dbl_bf[r * DBL_N + DT_RANK + D_STATE + s]);
    h = h * __expf(dt * Arow) + dt * uu * Bv;
    float yp = h * Cv;
    yp += __shfl_xor(yp, 1);
    yp += __shfl_xor(yp, 2);
    yp += __shfl_xor(yp, 4);
    yp += __shfl_xor(yp, 8);
    if (s == 0) {
      float z  = bf16_to_f32(xz_bf[r * (2 * D_INNER) + D_INNER + ch]);
      float sz = z / (1.f + __expf(-z));
      y_bf[r * D_INNER + ch] = f32_to_bf16((yp + uu * Dsk) * sz);
    }
  }
}

// ---------- host-side launcher ----------
extern "C" void kernel_launch(void* const* d_in, const int* in_sizes, int n_in,
                              void* d_out, int out_size, void* d_ws, size_t ws_size,
                              hipStream_t stream)
{
  (void)in_sizes; (void)n_in; (void)out_size; (void)ws_size;
  const float* x      = (const float*)d_in[0];
  const float* W_in   = (const float*)d_in[1];
  const float* conv_w = (const float*)d_in[2];
  const float* conv_b = (const float*)d_in[3];
  const float* W_xp   = (const float*)d_in[4];
  const float* W_dt   = (const float*)d_in[5];
  const float* b_dt   = (const float*)d_in[6];
  const float* A_log  = (const float*)d_in[7];
  const float* D_skip = (const float*)d_in[8];
  const float* W_out  = (const float*)d_in[9];

  // carve bf16 scratch from workspace
  unsigned short* p = (unsigned short*)d_ws;
  unsigned short* x_bf    = p; p += (size_t)MROWS * D_MODEL;
  unsigned short* Win_bf  = p; p += (size_t)2 * D_INNER * D_MODEL;
  unsigned short* Wxp_bf  = p; p += (size_t)DBL_N * D_INNER;
  unsigned short* Wdt_bf  = p; p += (size_t)D_INNER * DT_RANK;
  unsigned short* Wout_bf = p; p += (size_t)D_MODEL * D_INNER;
  unsigned short* xz_bf   = p; p += (size_t)MROWS * 2 * D_INNER;
  unsigned short* u_bf    = p; p += (size_t)MROWS * D_INNER;
  unsigned short* dbl_bf  = p; p += (size_t)MROWS * DBL_N;
  unsigned short* dtp_bf  = p; p += (size_t)MROWS * D_INNER;
  unsigned short* dt_bf   = p; p += (size_t)MROWS * D_INNER;
  unsigned short* y_bf    = p; p += (size_t)MROWS * D_INNER;

  dim3 blk(256);
  auto cvt = [&](const float* in, unsigned short* out, size_t n) {
    cvt_f32_bf16<<<dim3((unsigned)((n + 255) / 256)), blk, 0, stream>>>(in, out, (int)n);
  };
  cvt(x,     x_bf,    (size_t)MROWS * D_MODEL);
  cvt(W_in,  Win_bf,  (size_t)2 * D_INNER * D_MODEL);
  cvt(W_xp,  Wxp_bf,  (size_t)DBL_N * D_INNER);
  cvt(W_dt,  Wdt_bf,  (size_t)D_INNER * DT_RANK);
  cvt(W_out, Wout_bf, (size_t)D_MODEL * D_INNER);

  // GEMM1: xz = x * W_in^T   (8192x1024x4096) -> bf16
  gemm_bf16<<<dim3((2 * D_INNER) / BN, MROWS / BM), blk, 0, stream>>>(
      x_bf, D_MODEL, Win_bf, D_MODEL, xz_bf, 2 * D_INNER, D_MODEL, 2 * D_INNER, 1);

  // depthwise conv + bias + SiLU -> u
  conv_silu<<<dim3((MROWS * D_INNER) / 256), blk, 0, stream>>>(xz_bf, conv_w, conv_b, u_bf);

  // GEMM2: dbl = u * W_xproj^T  (8192x2048x96) -> bf16
  gemm_bf16<<<dim3((DBL_N + BN - 1) / BN, MROWS / BM), blk, 0, stream>>>(
      u_bf, D_INNER, Wxp_bf, D_INNER, dbl_bf, DBL_N, D_INNER, DBL_N, 1);

  // GEMM-dt: dt_pre = dbl[:, :64] * W_dt^T  (8192x64x2048), A row stride = 96
  gemm_bf16<<<dim3(D_INNER / BN, MROWS / BM), blk, 0, stream>>>(
      dbl_bf, DBL_N, Wdt_bf, DT_RANK, dtp_bf, D_INNER, DT_RANK, D_INNER, 1);

  // softplus(dt_pre + b_dt)
  dt_softplus<<<dim3((MROWS * D_INNER) / 256), blk, 0, stream>>>(dtp_bf, b_dt, dt_bf);

  // selective scan + skip + SiLU(z) gate -> y (bf16)
  scan_kernel<<<dim3(D_INNER / 16, BATCH), blk, 0, stream>>>(
      dt_bf, u_bf, dbl_bf, xz_bf, A_log, D_skip, y_bf);

  // GEMM3: out = y * W_out^T  (8192x2048x1024) -> f32 d_out
  gemm_bf16<<<dim3(D_MODEL / BN, MROWS / BM), blk, 0, stream>>>(
      y_bf, D_INNER, Wout_bf, D_INNER, d_out, D_MODEL, D_INNER, D_MODEL, 0);
}
